// MultiHeadGATLayer_11596411699835
// MI455X (gfx1250) — matrix operand order repaired and checked by reading, not compiled, not run on hardware
//
#include <hip/hip_runtime.h>
#include <hip/hip_bf16.h>
#include <stdint.h>

// ---------------------------------------------------------------------------
// MultiHead GAT layer, CDNA5 (gfx1250) WMMA implementation.
//   N=2048 nodes, F=512 in-features, H=8 heads, D=64 per head, alpha=0.2
// Pipeline:
//   k_convert  : h -> bf16 row-major ; W -> bf16 transposed (WT[n][k])
//   k_gemm_wh  : Wh = h@W + b via v_wmma_f32_16x16x32_bf16 ; also writes
//                WhT bf16 [512][2048] for the aggregation pass
//   k_srcdst   : src/dst logits per (node, head), head-major layout
//   k_attn     : fused masked-leakyrelu-softmax + WMMA aggregation,
//                M-blocked x2 (32 rows/wave) to halve B-operand L2 traffic
// ---------------------------------------------------------------------------

typedef __attribute__((ext_vector_type(16))) __bf16 v16bf;
typedef __attribute__((ext_vector_type(8)))  float  v8f;

union ABFrag {
    v16bf v;
    unsigned short u[16];
    uint4 q[2];
};
static_assert(sizeof(v16bf) == 32, "v16bf must be 8 VGPRs");

__device__ __forceinline__ unsigned short f2bf(float f) {
    unsigned int u = __float_as_uint(f);
    unsigned int r = (u + 0x7FFFu + ((u >> 16) & 1u)) >> 16;  // RNE
    return (unsigned short)r;
}

#define NNODES 2048
#define INF    512
#define NHEAD  8
#define DHEAD  64
#define ALPHA  0.2f

// --------------------------- kernel 0: convert ------------------------------
__global__ void k_convert(const float* __restrict__ h, const float* __restrict__ W,
                          unsigned short* __restrict__ h_bf,
                          unsigned short* __restrict__ WT_bf) {
    int t = blockIdx.x * 256 + threadIdx.x;
    if (t < NNODES * INF) h_bf[t] = f2bf(h[t]);
    if (t < INF * INF) {                 // W[k][n] -> WT[n][k]
        int k = t >> 9;
        int n = t & 511;
        WT_bf[(size_t)n * INF + k] = f2bf(W[t]);
    }
}

// --------------------------- kernel 1: Wh GEMM ------------------------------
// one wave per 16x16 output tile; 4 waves / block; 4096 tiles total.
__global__ void k_gemm_wh(const unsigned short* __restrict__ h_bf,
                          const unsigned short* __restrict__ WT_bf,
                          const float* __restrict__ W_b,
                          float* __restrict__ Wh,
                          unsigned short* __restrict__ WhT) {
    int lane = threadIdx.x & 31;
    int tile = blockIdx.x * 4 + (threadIdx.x >> 5);
    int i0 = (tile >> 5) << 4;          // 128 row tiles
    int n0 = (tile & 31) << 4;          // 32 col tiles
    int m  = lane & 15;
    int hi = lane >> 4;                 // lane half (K-interleave select)

    v8f acc = {};
    const unsigned short* arow = h_bf  + (size_t)(i0 + m) * INF + hi * 8;
    const unsigned short* brow = WT_bf + (size_t)(n0 + m) * INF + hi * 8;
    for (int kt = 0; kt < 16; ++kt) {
        ABFrag a, b;
        const uint4* ap = (const uint4*)(arow + kt * 32);
        a.q[0] = ap[0];                 // K = base .. base+7
        a.q[1] = ap[2];                 // K = base+16 .. base+23
        const uint4* bp = (const uint4*)(brow + kt * 32);
        b.q[0] = bp[0];                 // K = base .. base+7
        b.q[1] = bp[2];                 // K = base+16 .. base+23
        acc = __builtin_amdgcn_wmma_f32_16x16x32_bf16(
                  false, a.v, false, b.v, (short)0, acc, false, false);
    }
    float bias = W_b[n0 + m];
#pragma unroll
    for (int v = 0; v < 8; ++v) {
        int M = v + hi * 8;             // C/D layout: row = v + 8*half
        float val = acc[v] + bias;
        Wh [(size_t)(i0 + M) * INF + n0 + m] = val;
        WhT[(size_t)(n0 + m) * NNODES + i0 + M] = f2bf(val);
    }
}

// --------------------------- kernel 2: src/dst logits -----------------------
__global__ void k_srcdst(const float* __restrict__ Wh, const float* __restrict__ a_w,
                         float* __restrict__ src_t, float* __restrict__ dst_t) {
    int t = blockIdx.x * blockDim.x + threadIdx.x;
    if (t >= NNODES * NHEAD) return;
    int hd = t >> 11;
    int n  = t & (NNODES - 1);
    const float* row = Wh + (size_t)n * INF + hd * DHEAD;
    float s = 0.f, d = 0.f;
#pragma unroll 8
    for (int k = 0; k < DHEAD; ++k) {
        float w = row[k];
        s += w * a_w[k];
        d += w * a_w[DHEAD + k];
    }
    src_t[hd * NNODES + n] = s;
    dst_t[hd * NNODES + n] = d;
}

// --------------------------- kernel 3: fused attention ----------------------
// M-blocked x2: one wave per (32-row i-block, head) -> 512 waves.
// B fragments (WhT) are loaded once per j-chunk/d-tile and feed TWO WMMAs,
// halving B-operand L2 traffic vs. the unblocked version; dst_t values are
// shared between both row-tiles as well.
__global__ void k_attn(const int* __restrict__ adj,
                       const unsigned short* __restrict__ WhT,
                       const float* __restrict__ src_t,
                       const float* __restrict__ dst_t,
                       const float* __restrict__ a_b,
                       float* __restrict__ out) {
    int lane = threadIdx.x & 31;
    int gw   = blockIdx.x * 4 + (threadIdx.x >> 5);   // 0..511
    int ib   = gw >> 3;                               // 0..63 (32-row blocks)
    int hd   = gw & 7;
    int i0   = ib * 32;
    int m    = lane & 15;
    int hi   = lane >> 4;
    int iA   = i0 + m;                                // tile 0 row
    int iB   = i0 + 16 + m;                           // tile 1 row

    float s_iA = src_t[hd * NNODES + iA];
    float s_iB = src_t[hd * NNODES + iB];
    float ab   = a_b[0];
    const int baseA = hi * 8;                         // A-frag K base
    const int* adjA = adj + (size_t)iA * NNODES;
    const int* adjB = adj + (size_t)iB * NNODES;
    const float* drow = dst_t + hd * NNODES;
    const unsigned short* wb = WhT + (size_t)(hd * DHEAD) * NNODES;

    v8f acc0[4] = {v8f{}, v8f{}, v8f{}, v8f{}};
    v8f acc1[4] = {v8f{}, v8f{}, v8f{}, v8f{}};
    float denom0 = 0.f, denom1 = 0.f;

    for (int jt = 0; jt < NNODES / 32; ++jt) {
        int j0 = jt * 32;
        __builtin_prefetch(adjA + j0 + 32, 0, 1);     // global_prefetch next blocks
        __builtin_prefetch(adjB + j0 + 32, 0, 1);

        // dst contributions are identical for both row-tiles: load once.
        float dv[16];
#pragma unroll
        for (int e = 0; e < 16; ++e) {
            int j = j0 + baseA + (e & 7) + ((e >> 3) << 4);
            dv[e] = drow[j] + ab;
        }

        // Build both A fragments: p = exp(mask(leakyrelu(src_i + dst_j + b)))
        // Each lane's 16 elements are exactly its WMMA A-layout slots; the
        // K-runs of 8 are contiguous in j so adj loads vectorize to b128.
        ABFrag a0, a1;
#pragma unroll
        for (int e = 0; e < 16; ++e) {
            int j = j0 + baseA + (e & 7) + ((e >> 3) << 4);
            int eA = adjA[j];
            int eB = adjB[j];
            float s0 = s_iA + dv[e];
            float s1 = s_iB + dv[e];
            s0 = (s0 >= 0.f) ? s0 : ALPHA * s0;       // LeakyReLU
            s1 = (s1 >= 0.f) ? s1 : ALPHA * s1;
            s0 = (eA > 0) ? s0 : 0.f;                 // non-edge keeps score 0
            s1 = (eB > 0) ? s1 : 0.f;
            float p0 = __expf(s0);                    // exp(0)=1 still in denom
            float p1 = __expf(s1);
            denom0 += p0;
            denom1 += p1;
            a0.u[e] = f2bf(p0);
            a1.u[e] = f2bf(p1);
        }

        // Aggregate: each B fragment feeds both row-tiles (2x reuse).
#pragma unroll
        for (int t = 0; t < 4; ++t) {
            ABFrag b;
            const uint4* bp = (const uint4*)(wb + (size_t)(t * 16 + m) * NNODES
                                             + j0 + hi * 8);
            b.q[0] = bp[0];
            b.q[1] = bp[2];
            acc0[t] = __builtin_amdgcn_wmma_f32_16x16x32_bf16(
                          false, a0.v, false, b.v, (short)0, acc0[t], false, false);
            acc1[t] = __builtin_amdgcn_wmma_f32_16x16x32_bf16(
                          false, a1.v, false, b.v, (short)0, acc1[t], false, false);
        }
    }

    // Row denominators: lanes m and m+16 cover complementary K halves.
    denom0 += __shfl_xor(denom0, 16, 32);             // lane L: row (L&15) sum
    denom1 += __shfl_xor(denom1, 16, 32);

    // Normalize and store from C/D layout (M = v + 8*hi, N = m).
#pragma unroll
    for (int v = 0; v < 8; ++v) {
        int M = v + hi * 8;
        float inv0 = 1.f / __shfl(denom0, M, 32);
        float inv1 = 1.f / __shfl(denom1, M, 32);
#pragma unroll
        for (int t = 0; t < 4; ++t) {
            out[(size_t)(i0 + M) * (NHEAD * DHEAD) + hd * DHEAD + t * 16 + m] =
                acc0[t][v] * inv0;
            out[(size_t)(i0 + 16 + M) * (NHEAD * DHEAD) + hd * DHEAD + t * 16 + m] =
                acc1[t][v] * inv1;
        }
    }
}

// --------------------------- launcher ---------------------------------------
extern "C" void kernel_launch(void* const* d_in, const int* in_sizes, int n_in,
                              void* d_out, int out_size, void* d_ws, size_t ws_size,
                              hipStream_t stream) {
    const float* h   = (const float*)d_in[0];
    const int*   adj = (const int*)d_in[1];
    const float* W_w = (const float*)d_in[2];
    const float* W_b = (const float*)d_in[3];
    const float* a_w = (const float*)d_in[4];
    const float* a_b = (const float*)d_in[5];
    float* out = (float*)d_out;

    char* ws = (char*)d_ws;
    unsigned short* h_bf  = (unsigned short*)(ws);                 // 2 MB
    unsigned short* WT_bf = (unsigned short*)(ws + 2097152);       // 0.5 MB
    float*          Wh    = (float*)         (ws + 2621440);       // 4 MB
    unsigned short* WhT   = (unsigned short*)(ws + 6815744);       // 2 MB
    float*          src_t = (float*)         (ws + 8912896);       // 64 KB
    float*          dst_t = (float*)         (ws + 8978432);       // 64 KB
    (void)in_sizes; (void)n_in; (void)out_size; (void)ws_size;

    k_convert<<<dim3(4096), dim3(256), 0, stream>>>(h, W_w, h_bf, WT_bf);
    k_gemm_wh<<<dim3(1024), dim3(128), 0, stream>>>(h_bf, WT_bf, W_b, Wh, WhT);
    k_srcdst<<<dim3(64),   dim3(256), 0, stream>>>(Wh, a_w, src_t, dst_t);
    k_attn  <<<dim3(128),  dim3(128), 0, stream>>>(adj, WhT, src_t, dst_t, a_b, out);
}
